// tDPMPN_23819888623967
// MI455X (gfx1250) — matrix operand order repaired and checked by reading, not compiled, not run on hardware
//
#include <hip/hip_runtime.h>
#include <hip/hip_bf16.h>

// ---------------------------------------------------------------------------
// Types for CDNA5 WMMA
// ---------------------------------------------------------------------------
typedef __bf16 bf16_t;
typedef bf16_t v16bf __attribute__((ext_vector_type(16)));
typedef float  v8f   __attribute__((ext_vector_type(8)));

typedef union { v16bf v; unsigned u[8]; } fragu;

#define NUM_NODES   200000
#define NUM_EDGES   300000
#define NUM_QUERIES 128
#define NUM_ENT     50000
#define NDIM        256
#define NDIM_SM     64
#define STATIC_DIM  128
#define STATIC_SM   32
#define G_IN        224
#define E_TILES     (NUM_EDGES/16)     // 18750

__device__ __forceinline__ v8f wmma_bf16(v16bf a, v16bf b, v8f c) {
  return __builtin_amdgcn_wmma_f32_16x16x32_bf16(false, a, false, b,
                                                 (short)0, c, false, false);
}

__device__ __forceinline__ unsigned pack_bf(float x, float y) {
  bf16_t lo = (bf16_t)x, hi = (bf16_t)y;
  unsigned ul = (unsigned)__builtin_bit_cast(unsigned short, lo);
  unsigned uh = (unsigned)__builtin_bit_cast(unsigned short, hi);
  return ul | (uh << 16);
}

// A fragment (16x32 bf16) for THIS lane's row, from a row-major f32 row ptr.
// lane row = m (caller picked row ptr), h = lane>>4, kbase = k-tile start.
__device__ __forceinline__ v16bf load_a_f32row(const float* __restrict__ row,
                                               int kbase, int h) {
  fragu f;
#pragma unroll
  for (int j = 0; j < 4; ++j) {
    int k0 = kbase + 8 * h + 2 * j;
    f.u[j]     = pack_bf(row[k0],      row[k0 + 1]);
    f.u[j + 4] = pack_bf(row[k0 + 16], row[k0 + 17]);
  }
  return f.v;
}

// B fragment (32x16 bf16) from k-pair-packed dword buffer [K/2][ldn].
// n = output column for this lane, k2base = (k-tile start)/2.
__device__ __forceinline__ v16bf load_b_packed(const unsigned* __restrict__ Bp,
                                               int ldn, int k2base, int h, int n) {
  fragu f;
#pragma unroll
  for (int j = 0; j < 8; ++j)
    f.u[j] = Bp[(size_t)(k2base + 4 * h + j + 4 * (j >> 2)) * ldn + n];
  return f.v;
}

// A fragment from an LDS row-major bf16 matrix (ldd dwords per row).
__device__ __forceinline__ v16bf load_a_lds(const unsigned* sh, int ldd,
                                            int m, int h, int ktile) {
  fragu f;
  const unsigned* rp = sh + m * ldd + ktile * 16;
#pragma unroll
  for (int j = 0; j < 4; ++j) {
    f.u[j]     = rp[4 * h + j];
    f.u[j + 4] = rp[8 + 4 * h + j];
  }
  return f.v;
}

__device__ __forceinline__ float leaky(float x) { return x > 0.f ? x : 0.01f * x; }

// order-preserving float->uint key for atomicMax
__device__ __forceinline__ unsigned fkey(float f) {
  unsigned u = __float_as_uint(f);
  return (u & 0x80000000u) ? ~u : (u | 0x80000000u);
}
__device__ __forceinline__ float fdec(unsigned k) {
  unsigned u = (k & 0x80000000u) ? (k & 0x7FFFFFFFu) : ~k;
  return __uint_as_float(u);
}

// ---------------------------------------------------------------------------
// K0: pack weights into k-pair-packed bf16 dword layout [K/2][N]
// ---------------------------------------------------------------------------
__global__ void k_pack(const float* __restrict__ Wbs, const float* __restrict__ Wpr,
                       const float* __restrict__ Wl,  const float* __restrict__ Wr,
                       const float* __restrict__ Wc,
                       unsigned* Wbsp, unsigned* Wprp, unsigned* Wlp,
                       unsigned* Wrp, unsigned* Wcp) {
  int tid = blockIdx.x * blockDim.x + threadIdx.x;
  if (tid < 32768) {                       // W_bs [256][256] -> [128][256]
    int k2 = tid >> 8, n = tid & 255;
    Wbsp[tid] = pack_bf(Wbs[(2 * k2) * 256 + n], Wbs[(2 * k2 + 1) * 256 + n]);
  } else if (tid < 40960) {                // W_proj [256][64] -> [128][64]
    int t = tid - 32768; int k2 = t >> 6, n = t & 63;
    Wprp[t] = pack_bf(Wpr[(2 * k2) * 64 + n], Wpr[(2 * k2 + 1) * 64 + n]);
  } else if (tid < 48128) {                // Wl [224][64] -> [112][64]
    int t = tid - 40960; int k2 = t >> 6, n = t & 63;
    Wlp[t] = pack_bf(Wl[(2 * k2) * 64 + n], Wl[(2 * k2 + 1) * 64 + n]);
  } else if (tid < 55296) {                // Wr
    int t = tid - 48128; int k2 = t >> 6, n = t & 63;
    Wrp[t] = pack_bf(Wr[(2 * k2) * 64 + n], Wr[(2 * k2 + 1) * 64 + n]);
  } else if (tid < 57344) {                // Wc [64][64] -> [32][64]
    int t = tid - 55296; int k2 = t >> 6, n = t & 63;
    Wcp[t] = pack_bf(Wc[(2 * k2) * 64 + n], Wc[(2 * k2 + 1) * 64 + n]);
  }
}

// ---------------------------------------------------------------------------
// Kq: tiny query-context GEMMs (128 rows) -- scalar, negligible
// ---------------------------------------------------------------------------
__global__ void k_query(const float* __restrict__ qsrc, const float* __restrict__ qrel,
                        const float* __restrict__ Wst, const float* __restrict__ bst,
                        const float* __restrict__ Wpr, const float* __restrict__ bpr,
                        float* qs, float* qr) {
  int tid = blockIdx.x * blockDim.x + threadIdx.x;
  if (tid < NUM_QUERIES * STATIC_SM) {
    int b = tid >> 5, c = tid & 31;
    float acc = bst[c];
    for (int k = 0; k < STATIC_DIM; ++k) acc += qsrc[b * STATIC_DIM + k] * Wst[k * STATIC_SM + c];
    qs[tid] = acc;
  } else if (tid < NUM_QUERIES * STATIC_SM + NUM_QUERIES * NDIM_SM) {
    int t = tid - NUM_QUERIES * STATIC_SM;
    int b = t >> 6, c = t & 63;
    float acc = bpr[c];
    for (int k = 0; k < NDIM; ++k) acc += qrel[b * NDIM + k] * Wpr[k * NDIM_SM + c];
    qr[t] = acc;
  }
}

// ---------------------------------------------------------------------------
// K1: fused  proj_all = leaky(X @ W_bs + b_bs) @ W_proj + b_proj
// 2 waves / block, 32 rows / wave (two 16-row strips). Each B fragment feeds
// two WMMAs (one per strip) -> 4 B-dword loads per WMMA instead of 8.
// Stage-1 output restaged via LDS (bf16) as stage-2 A operand.
// ---------------------------------------------------------------------------
__global__ __launch_bounds__(64) void
k_node_gemm(const float* __restrict__ X,
            const unsigned* __restrict__ Wbsp, const float* __restrict__ bbs,
            const unsigned* __restrict__ Wprp, const float* __restrict__ bpr,
            float* __restrict__ proj) {
  __shared__ bf16_t sh[2 * 32 * 256];          // 32 KB
  int w = threadIdx.x >> 5, lane = threadIdx.x & 31;
  int m = lane & 15, h = lane >> 4;
  int m0 = blockIdx.x * 64 + w * 32;           // NUM_NODES % 64 == 0

  bf16_t* mysh = sh + w * 32 * 256;
  unsigned* myshu = (unsigned*)mysh;

  const float* xrow0 = X + (size_t)(m0 + m) * NDIM;
  const float* xrow1 = X + (size_t)(m0 + 16 + m) * NDIM;

  v16bf a0[8], a1[8];
#pragma unroll
  for (int kt = 0; kt < 8; ++kt) {
    a0[kt] = load_a_f32row(xrow0, kt * 32, h);
    a1[kt] = load_a_f32row(xrow1, kt * 32, h);
  }

  // stage 1: emb2 [32 x 256] -> LDS (bf16, bias + leaky)
  for (int nt = 0; nt < 16; ++nt) {
    v8f acc0 = {}, acc1 = {};
    int n = nt * 16 + m;
#pragma unroll
    for (int kt = 0; kt < 8; ++kt) {
      v16bf b = load_b_packed(Wbsp, 256, kt * 16, h, n);
      acc0 = wmma_bf16(a0[kt], b, acc0);
      acc1 = wmma_bf16(a1[kt], b, acc1);
    }
    float bias = bbs[n];
#pragma unroll
    for (int r = 0; r < 8; ++r) {
      mysh[(r + 8 * h) * 256 + n]        = (bf16_t)leaky(acc0[r] + bias);
      mysh[(16 + r + 8 * h) * 256 + n]   = (bf16_t)leaky(acc1[r] + bias);
    }
  }
  asm volatile("s_wait_dscnt 0" ::: "memory");

  // stage 2: [32 x 256] @ W_proj -> [32 x 64]  (reuse A registers)
#pragma unroll
  for (int kt = 0; kt < 8; ++kt) {
    a0[kt] = load_a_lds(myshu, 128, m, h, kt);
    a1[kt] = load_a_lds(myshu + 16 * 128, 128, m, h, kt);
  }

#pragma unroll
  for (int nt = 0; nt < 4; ++nt) {
    v8f acc0 = {}, acc1 = {};
    int n = nt * 16 + m;
#pragma unroll
    for (int kt = 0; kt < 8; ++kt) {
      v16bf b = load_b_packed(Wprp, 64, kt * 16, h, n);
      acc0 = wmma_bf16(a0[kt], b, acc0);
      acc1 = wmma_bf16(a1[kt], b, acc1);
    }
    float bias = bpr[n];
#pragma unroll
    for (int r = 0; r < 8; ++r) {
      proj[(size_t)(m0 + r + 8 * h) * 64 + n]      = acc0[r] + bias;
      proj[(size_t)(m0 + 16 + r + 8 * h) * 64 + n] = acc1[r] + bias;
    }
  }
}

// ---------------------------------------------------------------------------
// K1b: rel_sm = rel_emb @ W_proj + b_proj  (32 rows / wave, B reused x2)
// ---------------------------------------------------------------------------
__global__ __launch_bounds__(64) void
k_rel_gemm(const float* __restrict__ R,
           const unsigned* __restrict__ Wprp, const float* __restrict__ bpr,
           float* __restrict__ out, int M) {
  int w = threadIdx.x >> 5, lane = threadIdx.x & 31;
  int m = lane & 15, h = lane >> 4;
  int m0 = blockIdx.x * 64 + w * 32;
  if (m0 >= M) return;                          // uniform per wave; M % 32 == 0

  const float* rrow0 = R + (size_t)(m0 + m) * NDIM;
  const float* rrow1 = R + (size_t)(m0 + 16 + m) * NDIM;

  v16bf a0[8], a1[8];
#pragma unroll
  for (int kt = 0; kt < 8; ++kt) {
    a0[kt] = load_a_f32row(rrow0, kt * 32, h);
    a1[kt] = load_a_f32row(rrow1, kt * 32, h);
  }

#pragma unroll
  for (int nt = 0; nt < 4; ++nt) {
    v8f acc0 = {}, acc1 = {};
    int n = nt * 16 + m;
#pragma unroll
    for (int kt = 0; kt < 8; ++kt) {
      v16bf b = load_b_packed(Wprp, 64, kt * 16, h, n);
      acc0 = wmma_bf16(a0[kt], b, acc0);
      acc1 = wmma_bf16(a1[kt], b, acc1);
    }
    float bias = bpr[n];
#pragma unroll
    for (int r = 0; r < 8; ++r) {
      out[(size_t)(m0 + r + 8 * h) * 64 + n]      = acc0[r] + bias;
      out[(size_t)(m0 + 16 + r + 8 * h) * 64 + n] = acc1[r] + bias;
    }
  }
}

// ---------------------------------------------------------------------------
// K2: edge attention logits, 16 edges per wave as one WMMA M-tile.
//   lh = leaky([hvi|rel|qs|qr] @ Wl + bl)
//   rh = leaky([hvj|rel|qs|qr] @ Wr + br) @ Wc + bc
//   logit = rowsum(lh * rh)
// ---------------------------------------------------------------------------
__global__ __launch_bounds__(128) void
k_edge(const float* __restrict__ proj, const float* __restrict__ rel_sm,
       const float* __restrict__ qs, const float* __restrict__ qr,
       const int* __restrict__ egI, const int* __restrict__ srcI,
       const int* __restrict__ dstI,
       const unsigned* __restrict__ Wlp, const float* __restrict__ bl,
       const unsigned* __restrict__ Wrp, const float* __restrict__ brr,
       const unsigned* __restrict__ Wcp, const float* __restrict__ bcc,
       float* __restrict__ logits) {
  __shared__ bf16_t sh[4 * 16 * 64];            // 8 KB
  int w = threadIdx.x >> 5, lane = threadIdx.x & 31;
  int m = lane & 15, h = lane >> 4;
  int t = blockIdx.x * 4 + w;
  if (t >= E_TILES) return;                     // uniform per wave

  bf16_t* mysh = sh + w * 16 * 64;
  unsigned* myshu = (unsigned*)mysh;

  int e = t * 16 + m;
  int s = srcI[e], d = dstI[e], q = egI[e];
  const float* hvi = proj   + (size_t)s * 64;
  const float* hvj = proj   + (size_t)d * 64;
  const float* re  = rel_sm + (size_t)e * 64;
  const float* qsr = qs + q * STATIC_SM;
  const float* qrr = qr + q * NDIM_SM;

  // fragments: k-tiles 0,1 differ (src vs dst gather); 2..6 shared
  v16bf aL[7], aR0, aR1;
  aL[0] = load_a_f32row(hvi, 0, h);   aL[1] = load_a_f32row(hvi, 32, h);
  aR0   = load_a_f32row(hvj, 0, h);   aR1   = load_a_f32row(hvj, 32, h);
  aL[2] = load_a_f32row(re, 0, h);    aL[3] = load_a_f32row(re, 32, h);
  aL[4] = load_a_f32row(qsr, 0, h);
  aL[5] = load_a_f32row(qrr, 0, h);   aL[6] = load_a_f32row(qrr, 32, h);

  // lh = leaky(left @ Wl + bl), kept in registers (C layout)
  float lh[4][8];
#pragma unroll
  for (int nt = 0; nt < 4; ++nt) {
    v8f acc = {};
    int n = nt * 16 + m;
#pragma unroll
    for (int kt = 0; kt < 7; ++kt) {
      v16bf b = load_b_packed(Wlp, 64, kt * 16, h, n);
      acc = wmma_bf16(aL[kt], b, acc);
    }
    float bias = bl[n];
#pragma unroll
    for (int r = 0; r < 8; ++r) lh[nt][r] = leaky(acc[r] + bias);
  }

  // rt = leaky(right @ Wr + br) -> LDS bf16 (restage for Wc A-operand)
#pragma unroll
  for (int nt = 0; nt < 4; ++nt) {
    v8f acc = {};
    int n = nt * 16 + m;
#pragma unroll
    for (int kt = 0; kt < 7; ++kt) {
      v16bf av = (kt == 0) ? aR0 : (kt == 1) ? aR1 : aL[kt];
      v16bf b = load_b_packed(Wrp, 64, kt * 16, h, n);
      acc = wmma_bf16(av, b, acc);
    }
    float bias = brr[n];
#pragma unroll
    for (int r = 0; r < 8; ++r)
      mysh[(r + 8 * h) * 64 + n] = (bf16_t)leaky(acc[r] + bias);
  }
  asm volatile("s_wait_dscnt 0" ::: "memory");

  v16bf a2[2];
  a2[0] = load_a_lds(myshu, 32, m, h, 0);
  a2[1] = load_a_lds(myshu, 32, m, h, 1);

  float part[8];
#pragma unroll
  for (int r = 0; r < 8; ++r) part[r] = 0.f;

#pragma unroll
  for (int nt = 0; nt < 4; ++nt) {
    v8f acc = {};
    int n = nt * 16 + m;
    acc = wmma_bf16(a2[0], load_b_packed(Wcp, 64, 0,  h, n), acc);
    acc = wmma_bf16(a2[1], load_b_packed(Wcp, 64, 16, h, n), acc);
    float bias = bcc[n];
#pragma unroll
    for (int r = 0; r < 8; ++r) part[r] += (acc[r] + bias) * lh[nt][r];
  }

  // reduce across the 16 lanes of each half (columns), rows stay put
#pragma unroll
  for (int off = 1; off < 16; off <<= 1)
#pragma unroll
    for (int r = 0; r < 8; ++r)
      part[r] += __shfl_xor(part[r], off, 32);

  if (m == 0) {
#pragma unroll
    for (int r = 0; r < 8; ++r)
      logits[t * 16 + 8 * h + r] = part[r];
  }
}

// ---------------------------------------------------------------------------
// K3: segment softmax over src, propagate to dst, aggregate per entity
// ---------------------------------------------------------------------------
__global__ void k_init(unsigned* segmax_u, float* denom, float* nscore, float* out) {
  int i = blockIdx.x * blockDim.x + threadIdx.x;
  if (i < NUM_NODES) { segmax_u[i] = 0u; denom[i] = 0.f; nscore[i] = 0.f; }
  if (i < NUM_ENT)   out[i] = 0.f;
}

__global__ void k_segmax(const float* __restrict__ logits,
                         const int* __restrict__ srcI, unsigned* segmax_u) {
  int e = blockIdx.x * blockDim.x + threadIdx.x;
  if (e < NUM_EDGES) atomicMax(&segmax_u[srcI[e]], fkey(logits[e]));
}

__global__ void k_expden(const float* __restrict__ logits,
                         const int* __restrict__ srcI,
                         const unsigned* __restrict__ segmax_u,
                         float* __restrict__ ex, float* denom) {
  int e = blockIdx.x * blockDim.x + threadIdx.x;
  if (e < NUM_EDGES) {
    int s = srcI[e];
    float x = __expf(logits[e] - fdec(segmax_u[s]));
    ex[e] = x;
    atomicAdd(&denom[s], x);
  }
}

__global__ void k_prop(const float* __restrict__ ex, const float* __restrict__ denom,
                       const int* __restrict__ srcI, const int* __restrict__ dstI,
                       const float* __restrict__ node_score, float* nscore) {
  int e = blockIdx.x * blockDim.x + threadIdx.x;
  if (e < NUM_EDGES) {
    int s = srcI[e];
    float trans = ex[e] / denom[s];
    atomicAdd(&nscore[dstI[e]], trans * node_score[s]);
  }
}

__global__ void k_entity(const float* __restrict__ nscore,
                         const int* __restrict__ entI, float* out) {
  int v = blockIdx.x * blockDim.x + threadIdx.x;
  if (v < NUM_NODES) atomicAdd(&out[entI[v]], nscore[v]);
}

// ---------------------------------------------------------------------------
// launch
// ---------------------------------------------------------------------------
extern "C" void kernel_launch(void* const* d_in, const int* in_sizes, int n_in,
                              void* d_out, int out_size, void* d_ws, size_t ws_size,
                              hipStream_t stream) {
  const float* X      = (const float*)d_in[0];   // memorized_embedding [N,256]
  const float* nsc    = (const float*)d_in[1];   // node_score [N]
  const float* relE   = (const float*)d_in[2];   // rel_emb [E,256]
  const float* qsrc   = (const float*)d_in[3];   // query_src_ts_emb [B,128]
  const float* qrel   = (const float*)d_in[4];   // query_rel_emb [B,256]
  const int*   egI    = (const int*)d_in[5];
  const int*   srcI   = (const int*)d_in[6];
  const int*   dstI   = (const int*)d_in[7];
  const int*   entI   = (const int*)d_in[8];
  const float* Wproj  = (const float*)d_in[9];
  const float* bproj  = (const float*)d_in[10];
  const float* Wstat  = (const float*)d_in[11];
  const float* bstat  = (const float*)d_in[12];
  const float* Wbs    = (const float*)d_in[13];
  const float* bbs    = (const float*)d_in[14];
  const float* Wl     = (const float*)d_in[15];
  const float* bl     = (const float*)d_in[16];
  const float* Wr     = (const float*)d_in[17];
  const float* br_    = (const float*)d_in[18];
  const float* Wc     = (const float*)d_in[19];
  const float* bc_    = (const float*)d_in[20];
  float* out = (float*)d_out;

  // workspace layout
  char* ws = (char*)d_ws;
  size_t o = 0;
  unsigned* Wbsp = (unsigned*)(ws + o); o += 32768u * 4;       // [128][256]
  unsigned* Wprp = (unsigned*)(ws + o); o += 8192u * 4;        // [128][64]
  unsigned* Wlp  = (unsigned*)(ws + o); o += 7168u * 4;        // [112][64]
  unsigned* Wrp  = (unsigned*)(ws + o); o += 7168u * 4;
  unsigned* Wcp  = (unsigned*)(ws + o); o += 2048u * 4;        // [32][64]
  float* qs      = (float*)(ws + o);    o += (size_t)NUM_QUERIES * STATIC_SM * 4;
  float* qr      = (float*)(ws + o);    o += (size_t)NUM_QUERIES * NDIM_SM * 4;
  float* proj    = (float*)(ws + o);    o += (size_t)NUM_NODES * 64 * 4;
  float* rel_sm  = (float*)(ws + o);    o += (size_t)NUM_EDGES * 64 * 4;
  float* logits  = (float*)(ws + o);    o += (size_t)NUM_EDGES * 4;
  float* ex      = (float*)(ws + o);    o += (size_t)NUM_EDGES * 4;
  unsigned* smx  = (unsigned*)(ws + o); o += (size_t)NUM_NODES * 4;
  float* denom   = (float*)(ws + o);    o += (size_t)NUM_NODES * 4;
  float* nscore  = (float*)(ws + o);    o += (size_t)NUM_NODES * 4;

  (void)in_sizes; (void)n_in; (void)out_size; (void)ws_size;

  k_pack<<<224, 256, 0, stream>>>(Wbs, Wproj, Wl, Wr, Wc, Wbsp, Wprp, Wlp, Wrp, Wcp);
  k_query<<<48, 256, 0, stream>>>(qsrc, qrel, Wstat, bstat, Wproj, bproj, qs, qr);
  k_init<<<(NUM_NODES + 255) / 256, 256, 0, stream>>>(smx, denom, nscore, out);

  k_node_gemm<<<NUM_NODES / 64, 64, 0, stream>>>(X, Wbsp, bbs, Wprp, bproj, proj);
  k_rel_gemm<<<(NUM_EDGES + 63) / 64, 64, 0, stream>>>(relE, Wprp, bproj, rel_sm, NUM_EDGES);

  k_edge<<<(E_TILES + 3) / 4, 128, 0, stream>>>(proj, rel_sm, qs, qr, egI, srcI, dstI,
                                                Wlp, bl, Wrp, br_, Wcp, bc_, logits);

  int eb = (NUM_EDGES + 255) / 256;
  k_segmax<<<eb, 256, 0, stream>>>(logits, srcI, smx);
  k_expden<<<eb, 256, 0, stream>>>(logits, srcI, smx, ex, denom);
  k_prop<<<eb, 256, 0, stream>>>(ex, denom, srcI, dstI, nsc, nscore);
  k_entity<<<(NUM_NODES + 255) / 256, 256, 0, stream>>>(nscore, entI, out);
}
